// MultiHeadAttention_12489764896825
// MI455X (gfx1250) — compile-verified, matrix operand order checked by reading, not run on hardware
//
#include <hip/hip_runtime.h>
#include <hip/hip_bf16.h>
#include <math.h>

// ---------------- problem constants (match reference) ----------------
constexpr int Hh  = 32;    // query heads
constexpr int KVH = 8;     // kv heads
constexpr int DH  = 64;    // head dim
constexpr int Dm  = 2048;  // model dim
constexpr int KVD = 2 * DH * KVH; // 1024

typedef __attribute__((ext_vector_type(16))) __bf16 v16bf;
typedef __attribute__((ext_vector_type(8)))  __bf16 v8bf;
typedef __attribute__((ext_vector_type(8)))  float  v8f;
typedef __attribute__((ext_vector_type(4)))  unsigned int u32x4;
typedef __attribute__((ext_vector_type(8)))  int  i32x8;
typedef __attribute__((ext_vector_type(4)))  int  i32x4;

// 16-bit fragment (16x32) per CDNA5 ISA 7.12.2: lane group "half" selects
// K = {half*8 .. half*8+7} (vgprs 0..3) and {16+half*8 ..+7} (vgprs 4..7).
// Each half-fragment is 8 contiguous bf16 = one 128-bit load.
__device__ __forceinline__ v16bf load_frag_k(const __bf16* rowptr, int k0, int half) {
  v8bf lo = *(const v8bf*)(rowptr + k0 + half * 8);
  v8bf hi = *(const v8bf*)(rowptr + k0 + 16 + half * 8);
  v16bf r;
  #pragma unroll
  for (int i = 0; i < 8; ++i) { r[i] = lo[i]; r[8 + i] = hi[i]; }
  return r;
}

// ---- TDM: issue tensor_load_to_lds of a [rows x 32] bf16 2D tile ----
// D# layout per CDNA5 ISA 8.3/8.4; 2D tensor -> groups 2/3 zero.
// This toolchain exposes the 6-arg builtin: (u32x4, i32x8, i32x4, i32x4, i32x8, cpol)
__device__ __forceinline__ void tdm_load_tile_bf16(
    unsigned lds_addr, const __bf16* gptr, unsigned k_len /*tensor_dim0*/,
    unsigned rows_total /*tensor_dim1*/, unsigned tile_k, unsigned tile_rows,
    unsigned long long stride0 /*elements*/) {
  unsigned long long ga = (unsigned long long)(size_t)gptr;
  u32x4 g0;
  g0[0] = 1u;                                   // count=1, user desc
  g0[1] = lds_addr;                             // LDS byte address
  g0[2] = (unsigned)(ga & 0xFFFFFFFFu);
  g0[3] = (unsigned)((ga >> 32) & 0x01FFFFFFu) | (2u << 30);  // type=2 (image)
  i32x8 g1;
  g1[0] = (int)(1u << 16);                      // wg_mask=0, data_size=1 (2 bytes)
  g1[1] = (int)((k_len & 0xFFFFu) << 16);       // tensor_dim0 lo16 at bit48
  g1[2] = (int)(((k_len >> 16) & 0xFFFFu) | ((rows_total & 0xFFFFu) << 16));
  g1[3] = (int)(((rows_total >> 16) & 0xFFFFu) | ((tile_k & 0xFFFFu) << 16));
  g1[4] = (int)(tile_rows & 0xFFFFu);           // tile_dim1; tile_dim2=0
  g1[5] = (int)(unsigned)(stride0 & 0xFFFFFFFFull);
  g1[6] = (int)(unsigned)((stride0 >> 32) & 0xFFFFull);  // stride1=0
  g1[7] = 0;
  i32x4 gz4 = {0, 0, 0, 0};
  i32x8 gz8 = {0, 0, 0, 0, 0, 0, 0, 0};
  __builtin_amdgcn_tensor_load_to_lds(g0, g1, gz4, gz4, gz8, 0);
}

// ---------------- 1) RMSNorm + cast to bf16 ----------------
__global__ __launch_bounds__(256)
void rmsnorm_bf16(const float* __restrict__ x, const float* __restrict__ w,
                  __bf16* __restrict__ xn, int D) {
  __shared__ float red[256];
  int row = blockIdx.x, t = threadIdx.x;
  const float* xr = x + (size_t)row * D;
  float ss = 0.f;
  for (int i = t; i < D; i += 256) { float v = xr[i]; ss += v * v; }
  red[t] = ss; __syncthreads();
  for (int s = 128; s > 0; s >>= 1) {
    if (t < s) red[t] += red[t + s];
    __syncthreads();
  }
  float r = rsqrtf(red[0] / (float)D + 1.1920929e-7f);
  __bf16* xo = xn + (size_t)row * D;
  for (int i = t; i < D; i += 256) xo[i] = (__bf16)(xr[i] * r * w[i]);
}

// ---------------- 1b) weight transpose + cast: Wt[n][k] = (bf16)W[k][n] --------
__global__ __launch_bounds__(256)
void w_transpose_cast(const float* __restrict__ W, __bf16* __restrict__ Wt,
                      int K, int Nout) {
  __shared__ float t[32][33];
  int n0 = blockIdx.x * 32, k0 = blockIdx.y * 32;
  int tx = threadIdx.x, ty = threadIdx.y;   // 32 x 8
  #pragma unroll
  for (int i = 0; i < 4; ++i)
    t[ty + 8 * i][tx] = W[(size_t)(k0 + ty + 8 * i) * Nout + n0 + tx];
  __syncthreads();
  #pragma unroll
  for (int i = 0; i < 4; ++i)
    Wt[(size_t)(n0 + ty + 8 * i) * K + k0 + tx] = (__bf16)t[tx][ty + 8 * i];
}

// ---------------- 2) GEMM: C[M,Nout] = A(bf16)[M,K] @ Wt(bf16)[Nout,K]^T + bias ----
// one wave computes a 32x64 tile: 2 M-frags x 4 N-frags, K-step 32.
// B slab (64 rows x 32 k) staged into LDS by the Tensor Data Mover,
// double-buffered on TENSORcnt; A streams straight from global (b128).
__global__ __launch_bounds__(32)
void gemm_bf16(const __bf16* __restrict__ A, const __bf16* __restrict__ Bt,
               const float* __restrict__ bias, float* __restrict__ C,
               int M, int K, int Nout) {
  __shared__ alignas(16) __bf16 ldsB[2][64 * 32];   // 2 x 4KB
  const int tn = blockIdx.x * 64;
  const int tm = blockIdx.y * 32;
  const int lane = threadIdx.x;
  const int m = lane & 15, half = lane >> 4;

  v8f zero = {0.f,0.f,0.f,0.f,0.f,0.f,0.f,0.f};
  v8f acc[2][4];
  #pragma unroll
  for (int i = 0; i < 2; ++i)
    #pragma unroll
    for (int j = 0; j < 4; ++j) acc[i][j] = zero;

  const __bf16* arow0 = A + (size_t)(tm + m) * K;
  const __bf16* arow1 = A + (size_t)(tm + 16 + m) * K;
  const __bf16* bbase = Bt + (size_t)tn * K;        // 64-row slab base

  // raw LDS byte offsets (low 32 bits of the LDS aperture address)
  unsigned ldsOff[2];
  ldsOff[0] = (unsigned)(size_t)&ldsB[0][0];
  ldsOff[1] = (unsigned)(size_t)&ldsB[1][0];

  // preload first B slab
  tdm_load_tile_bf16(ldsOff[0], bbase, (unsigned)K, 64u, 32u, 64u,
                     (unsigned long long)K);

  int buf = 0;
  for (int k0 = 0; k0 < K; k0 += 32) {
    if (k0 + 32 < K) {
      // WAR guard: prior ds reads of buf^1 must retire before TDM overwrites it
      asm volatile("s_wait_dscnt 0x0" ::: "memory");
      tdm_load_tile_bf16(ldsOff[buf ^ 1], bbase + (k0 + 32), (unsigned)K, 64u,
                         32u, 64u, (unsigned long long)K);
      __builtin_amdgcn_s_wait_tensorcnt(1);   // current slab ready
    } else {
      __builtin_amdgcn_s_wait_tensorcnt(0);
    }

    v16bf a0 = load_frag_k(arow0, k0, half);
    v16bf a1 = load_frag_k(arow1, k0, half);
    const __bf16* bl = &ldsB[buf][0];
    #pragma unroll
    for (int j = 0; j < 4; ++j) {
      v16bf b = load_frag_k(bl + (j * 16 + m) * 32, 0, half);
      acc[0][j] = __builtin_amdgcn_wmma_f32_16x16x32_bf16(
          false, a0, false, b, (short)0, acc[0][j], false, false);
      acc[1][j] = __builtin_amdgcn_wmma_f32_16x16x32_bf16(
          false, a1, false, b, (short)0, acc[1][j], false, false);
    }
    buf ^= 1;
  }
  #pragma unroll
  for (int i = 0; i < 2; ++i)
    #pragma unroll
    for (int j = 0; j < 4; ++j)
      #pragma unroll
      for (int v = 0; v < 8; ++v) {
        int row = tm + i * 16 + v + 8 * half;
        int col = tn + j * 16 + m;
        C[(size_t)row * Nout + col] = acc[i][j][v] + bias[col];
      }
}

// ---------------- 3a) RoPE on Q + repack head-major bf16 ----------------
__global__ __launch_bounds__(32)
void rope_q_pack(const float* __restrict__ qf, __bf16* __restrict__ qb,
                 const int* __restrict__ posp, int N) {
  int n = blockIdx.x, h = blockIdx.y, i = threadIdx.x;     // pair index 0..31
  float freq = __expf(-logf(10000.f) * (2.f * i) / (float)DH);
  float ang = (float)(posp[0] + n) * freq;
  float c = __cosf(ang), s = __sinf(ang);
  const float* src = qf + (size_t)n * Dm + h * DH;
  float x1 = src[2 * i], x2 = src[2 * i + 1];
  __bf16* dst = qb + ((size_t)h * N + n) * DH;
  dst[2 * i]     = (__bf16)(x1 * c - x2 * s);
  dst[2 * i + 1] = (__bf16)(x2 * c + x1 * s);
}

// ------- 3b) RoPE on K, split K/V; K head-major, V transposed [kvh][DH][N] -------
__global__ __launch_bounds__(32)
void rope_kv_pack(const float* __restrict__ kvf, __bf16* __restrict__ kb,
                  __bf16* __restrict__ vT, int N) {
  int n = blockIdx.x, kvh = blockIdx.y, i = threadIdx.x;   // pair index 0..31
  const float* src = kvf + (size_t)n * KVD + kvh * 2 * DH;
  float freq = __expf(-logf(10000.f) * (2.f * i) / (float)DH);
  float ang = (float)n * freq;                              // k positions start at 0
  float c = __cosf(ang), s = __sinf(ang);
  float x1 = src[2 * i], x2 = src[2 * i + 1];
  __bf16* kd = kb + ((size_t)kvh * N + n) * DH;
  kd[2 * i]     = (__bf16)(x1 * c - x2 * s);
  kd[2 * i + 1] = (__bf16)(x2 * c + x1 * s);
  // transposed V: vT[kvh][dh][n]
  vT[((size_t)kvh * DH + 2 * i)     * N + n] = (__bf16)src[DH + 2 * i];
  vT[((size_t)kvh * DH + 2 * i + 1) * N + n] = (__bf16)src[DH + 2 * i + 1];
}

// ---------------- 4) Flash-style causal attention, one wave per (head, 16-q tile) ---
__global__ __launch_bounds__(32)
void attn_wmma(const __bf16* __restrict__ qb, const __bf16* __restrict__ kbuf,
               const __bf16* __restrict__ vT, __bf16* __restrict__ ob, int N) {
  __shared__ float  sS[16][33];                   // raw scores for a 32-key pair
  __shared__ alignas(16) __bf16 sP[16][32];       // exp probabilities (bf16)
  __shared__ float  sM[16], sL[16], sCorr[16];

  const int qt = blockIdx.x;        // 16-query tile
  const int h  = blockIdx.y;
  const int kvh = h / (Hh / KVH);
  const int lane = threadIdx.x;
  const int m = lane & 15, half = lane >> 4;
  const float scale = 0.125f;       // 1/sqrt(DH)

  if (lane < 16) { sM[lane] = -INFINITY; sL[lane] = 0.f; }
  __syncthreads();

  // Q A-fragments for the two 32-wide dh chunks (loaded once)
  const __bf16* qrow = qb + ((size_t)h * N + (size_t)qt * 16 + m) * DH;
  v16bf qa[2];
  #pragma unroll
  for (int c = 0; c < 2; ++c) qa[c] = load_frag_k(qrow, c * 32, half);

  v8f zero = {0.f,0.f,0.f,0.f,0.f,0.f,0.f,0.f};
  v8f acc[4];
  #pragma unroll
  for (int j = 0; j < 4; ++j) acc[j] = zero;

  const __bf16* kh = kbuf + (size_t)kvh * N * DH;
  const __bf16* vh = vT   + (size_t)kvh * DH * N;
  const int lastkey = qt * 16 + 15;
  const int npairs = lastkey / 32 + 1;

  for (int p = 0; p < npairs; ++p) {
    // ---- S = Q K^T for the two 16-key sub-tiles of this pair ----
    #pragma unroll
    for (int sub = 0; sub < 2; ++sub) {
      int kb0 = p * 32 + sub * 16;
      const __bf16* krow = kh + (size_t)(kb0 + m) * DH;
      v8f s = zero;
      #pragma unroll
      for (int c = 0; c < 2; ++c) {
        v16bf bf = load_frag_k(krow, c * 32, half);
        s = __builtin_amdgcn_wmma_f32_16x16x32_bf16(
            false, qa[c], false, bf, (short)0, s, false, false);
      }
      #pragma unroll
      for (int v = 0; v < 8; ++v) sS[v + 8 * half][sub * 16 + m] = s[v];
    }
    __syncthreads();

    // ---- online softmax over the 32 keys (lanes 0..15 own one row each) ----
    if (lane < 16) {
      int r = lane;
      int qg = qt * 16 + r;
      float mold = sM[r];
      float mx = mold;
      float vals[32];
      #pragma unroll
      for (int c = 0; c < 32; ++c) {
        int kg = p * 32 + c;
        float xv = (kg <= qg) ? sS[r][c] * scale : -INFINITY;
        vals[c] = xv;
        mx = fmaxf(mx, xv);
      }
      float corr, lsum = 0.f;
      if (mx == -INFINITY) {          // fully masked pair for this row
        corr = 1.f;
        #pragma unroll
        for (int c = 0; c < 32; ++c) sP[r][c] = (__bf16)0.f;
      } else {
        float dm = mold - mx;
        corr = (mold == -INFINITY) ? 0.f : __expf(dm);
        #pragma unroll
        for (int c = 0; c < 32; ++c) {
          float e = (vals[c] == -INFINITY) ? 0.f : __expf(vals[c] - mx);
          sP[r][c] = (__bf16)e;
          lsum += e;
        }
      }
      sM[r] = mx;
      sL[r] = sL[r] * corr + lsum;
      sCorr[r] = corr;
    }
    __syncthreads();

    // ---- rescale running O accumulators ----
    #pragma unroll
    for (int v = 0; v < 8; ++v) {
      float cr = sCorr[v + 8 * half];
      acc[0][v] *= cr; acc[1][v] *= cr; acc[2][v] *= cr; acc[3][v] *= cr;
    }

    // ---- P A-fragment (16x32) from LDS: two ds_load_b128 ----
    v16bf pa = load_frag_k(&sP[m][0], 0, half);

    // ---- O += P @ V for the 4 dh sub-tiles (V is [dh][key]: K-contiguous) ----
    #pragma unroll
    for (int j = 0; j < 4; ++j) {
      const __bf16* vrow = vh + (size_t)(j * 16 + m) * N;
      v16bf vf = load_frag_k(vrow, p * 32, half);
      acc[j] = __builtin_amdgcn_wmma_f32_16x16x32_bf16(
          false, pa, false, vf, (short)0, acc[j], false, false);
    }
    __syncthreads();   // protect sS/sP for next pair
  }

  // ---- normalize and write O (token-major bf16, ready for O-projection GEMM) ----
  #pragma unroll
  for (int v = 0; v < 8; ++v) {
    int r = v + 8 * half;
    float linv = 1.f / sL[r];
    int tok = qt * 16 + r;
    __bf16* od = ob + (size_t)tok * Dm + h * DH;
    #pragma unroll
    for (int j = 0; j < 4; ++j)
      od[j * 16 + m] = (__bf16)(acc[j][v] * linv);
  }
}

// ---------------- launcher ----------------
extern "C" void kernel_launch(void* const* d_in, const int* in_sizes, int n_in,
                              void* d_out, int out_size, void* d_ws, size_t ws_size,
                              hipStream_t stream) {
  const float* x      = (const float*)d_in[0];
  const float* norm_w = (const float*)d_in[1];
  const float* Wq     = (const float*)d_in[2];
  const float* bq     = (const float*)d_in[3];
  const float* Wkv    = (const float*)d_in[4];
  const float* bkv    = (const float*)d_in[5];
  const float* Wo     = (const float*)d_in[6];
  const float* bo     = (const float*)d_in[7];
  const int*   posp   = (const int*)d_in[8];
  float* out = (float*)d_out;

  const int N = in_sizes[0] / Dm;   // tokens (B=1)

  // workspace carve-up
  char* ws = (char*)d_ws;
  size_t off = 0;
  auto carve = [&](size_t bytes) { size_t o = off; off = (off + bytes + 255) & ~(size_t)255; return o; };
  __bf16* xn_bf = (__bf16*)(ws + carve((size_t)N * Dm  * sizeof(__bf16)));
  float*  qf    = (float*) (ws + carve((size_t)N * Dm  * sizeof(float)));
  float*  kvf   = (float*) (ws + carve((size_t)N * KVD * sizeof(float)));
  __bf16* q_bf  = (__bf16*)(ws + carve((size_t)N * Dm  * sizeof(__bf16)));
  __bf16* k_bf  = (__bf16*)(ws + carve((size_t)KVH * N * DH * sizeof(__bf16)));
  __bf16* v_bfT = (__bf16*)(ws + carve((size_t)KVH * DH * N * sizeof(__bf16)));
  __bf16* o_bf  = (__bf16*)(ws + carve((size_t)N * Dm  * sizeof(__bf16)));
  __bf16* WqT   = (__bf16*)(ws + carve((size_t)Dm * Dm  * sizeof(__bf16)));
  __bf16* WkvT  = (__bf16*)(ws + carve((size_t)KVD * Dm * sizeof(__bf16)));
  __bf16* WoT   = (__bf16*)(ws + carve((size_t)Dm * Dm  * sizeof(__bf16)));
  (void)ws_size; (void)n_in; (void)out_size;

  // 0) weights: transpose + cast to bf16 (Wt[n][k], K-contiguous fragments)
  w_transpose_cast<<<dim3(Dm  / 32, Dm / 32), dim3(32, 8), 0, stream>>>(Wq,  WqT,  Dm, Dm);
  w_transpose_cast<<<dim3(KVD / 32, Dm / 32), dim3(32, 8), 0, stream>>>(Wkv, WkvT, Dm, KVD);
  w_transpose_cast<<<dim3(Dm  / 32, Dm / 32), dim3(32, 8), 0, stream>>>(Wo,  WoT,  Dm, Dm);

  // 1) RMSNorm -> bf16
  rmsnorm_bf16<<<N, 256, 0, stream>>>(x, norm_w, xn_bf, Dm);

  // 2) Q and KV projections (WMMA + TDM-staged B)
  gemm_bf16<<<dim3(Dm / 64,  N / 32), 32, 0, stream>>>(xn_bf, WqT,  bq,  qf,  N, Dm, Dm);
  gemm_bf16<<<dim3(KVD / 64, N / 32), 32, 0, stream>>>(xn_bf, WkvT, bkv, kvf, N, Dm, KVD);

  // 3) RoPE + repack
  rope_q_pack <<<dim3(N, Hh),  32, 0, stream>>>(qf,  q_bf, posp, N);
  rope_kv_pack<<<dim3(N, KVH), 32, 0, stream>>>(kvf, k_bf, v_bfT, N);

  // 4) causal attention (WMMA)
  attn_wmma<<<dim3(N / 16, Hh), 32, 0, stream>>>(q_bf, k_bf, v_bfT, o_bf, N);

  // 5) output projection (WMMA + TDM-staged B) -> f32 out
  gemm_bf16<<<dim3(Dm / 64, N / 32), 32, 0, stream>>>(o_bf, WoT, bo, out, N, Dm, Dm);
}
